// GNN_5866925326813
// MI455X (gfx1250) — matrix operand order repaired and checked by reading, not compiled, not run on hardware
//
#include <hip/hip_runtime.h>
#include <hip/hip_bf16.h>
#include <stdint.h>

typedef __bf16 bf16;
typedef __attribute__((ext_vector_type(16))) __bf16 v16bf;
typedef __attribute__((ext_vector_type(8)))  float  v8f;
typedef __attribute__((ext_vector_type(4)))  int    v4i;

// ---------------------------------------------------------------------------
// WMMA helpers (CDNA5 wave32, 16x16x32 bf16, f32 accumulate)
// A layout (16x32, 16-bit): lanes 0-15 -> M=lane, K 0..15
//                           lanes 16-31 -> M=lane-16, K 16..31
// B layout (32x16, 16-bit): lanes 0-15 -> N=lane, K 0..15
//                           lanes 16-31 -> N=lane-16, K 16..31
// A tiles staged in LDS pre-swizzled in fragment order: for K-block kb, lane
// l's 16 elements live at As[kb*512 + l*16 .. +15] -> one aligned 32B LDS read.
// B stored transposed [Nout][Kpad] -> one contiguous 32B global load.
// ---------------------------------------------------------------------------
static __device__ __forceinline__ v16bf load_b_frag(const bf16* Bt, int pitch,
                                                    int lane, int k0) {
  const bf16* p = Bt + (size_t)(lane & 15) * pitch + k0 + ((lane >> 4) << 4);
  return *(const v16bf*)p;
}

static __device__ __forceinline__ v8f wmma_bf16(v16bf a, v16bf b, v8f c) {
  return __builtin_amdgcn_wmma_f32_16x16x32_bf16(false, a, false, b,
                                                 (short)0, c, false, false);
}

// ---- gfx1250 async global->LDS copy (ASYNCcnt path), with safe fallback ----
#if defined(__has_builtin)
#if __has_builtin(__builtin_amdgcn_global_load_async_to_lds_b128)
#define GNN_ASYNC_LDS 1
#endif
#endif

static __device__ __forceinline__ void async_copy_b128(const uint4* g, bf16* l) {
#ifdef GNN_ASYNC_LDS
  // builtin signature (from hipcc diagnostic): (v4i addrspace(1)*, v4i addrspace(3)*, imm, imm)
  typedef __attribute__((address_space(1))) v4i gv4i;
  typedef __attribute__((address_space(3))) v4i lv4i;
  __builtin_amdgcn_global_load_async_to_lds_b128(
      (gv4i*)(uintptr_t)g, (lv4i*)(uint32_t)(uintptr_t)l, 0, 0);
#else
  *(uint4*)l = *g;
#endif
}

static __device__ __forceinline__ void async_join() {
#ifdef GNN_ASYNC_LDS
#if __has_builtin(__builtin_amdgcn_s_wait_asynccnt)
  __builtin_amdgcn_s_wait_asynccnt(0);
#else
  asm volatile("s_wait_asynccnt 0" ::: "memory");
#endif
#endif
}

static __device__ __forceinline__ void atomicMaxF(float* addr, float val) {
  int* ai = (int*)addr;
  int old = __float_as_int(*addr);
  while (__int_as_float(old) < val) {
    int assumed = old;
    old = atomicCAS(ai, assumed, __float_as_int(val));
    if (old == assumed) break;
  }
}

// ---------------------------------------------------------------------------
// Weight conversion: W f32 [K][N] -> Wt bf16 [Npad][Kpad] (transposed, padded)
// ---------------------------------------------------------------------------
__global__ void gnn_convW(const float* __restrict__ W, int K, int N, int Kpad,
                          bf16* __restrict__ out, int tot) {
  int idx = blockIdx.x * blockDim.x + threadIdx.x;
  if (idx >= tot) return;
  int nn = idx / Kpad, k = idx - nn * Kpad;
  out[idx] = (k < K && nn < N) ? (bf16)W[(size_t)k * N + nn] : (bf16)0.f;
}

// Pack [X1 | X2 | 0] rows into bf16 A buffer [nPad][Kpad]. X1 pitch K1, X2 pitch 256.
__global__ void gnn_pack(const float* __restrict__ X1, int K1,
                         const float* __restrict__ X2, int K2,
                         int n, int Kpad, bf16* __restrict__ out, int tot) {
  int idx = blockIdx.x * blockDim.x + threadIdx.x;
  if (idx >= tot) return;
  int r = idx / Kpad, k = idx - r * Kpad;
  float v = 0.f;
  if (r < n) {
    if (k < K1)            v = X1[(size_t)r * K1 + k];
    else if (k < K1 + K2)  v = X2[(size_t)r * 256 + (k - K1)];
  }
  out[idx] = (bf16)v;
}

// ---------------------------------------------------------------------------
// Generic WMMA GEMM: C[n][ldc] = A_bf16[nPad][Kpad] * Bt_bf16[NoutPad][Kpad]
// block = 256 threads = 8 waves; block tile = 16 rows x NoutPad cols
// each wave: 2 col tiles; B fragments software-pipelined one K-step ahead
// ---------------------------------------------------------------------------
__global__ void gnn_gemm(const bf16* __restrict__ A, int Kpad,
                         const bf16* __restrict__ Bt, int NoutPad,
                         const float* __restrict__ bias,
                         float* __restrict__ C, int n, int ldc, int relu) {
  __shared__ __attribute__((aligned(32))) bf16 As[16 * 512];  // Kpad <= 512
  const int tid = threadIdx.x, lane = tid & 31;
  const int wave = __builtin_amdgcn_readfirstlane(tid >> 5);  // provably scalar
  const int m0 = blockIdx.x * 16;

  // cooperative async load, swizzling into fragment order (8-elem aligned
  // chunks stay contiguous under the swizzle -> b128 copies are legal)
  const int vecsPerRow = Kpad >> 3;
  const int nvec = 16 * vecsPerRow;
  const uint4* Ag = (const uint4*)(A + (size_t)m0 * Kpad);
  for (int i = tid; i < nvec; i += blockDim.x) {
    int row = i / vecsPerRow;
    int kv  = (i - row * vecsPerRow) << 3;
    int kb  = kv >> 5, kr = kv & 31;
    int l   = row + ((kr >> 4) << 4);
    int el  = kr & 15;
    async_copy_b128(Ag + i, As + kb * 512 + l * 16 + el);
  }
  async_join();
  __syncthreads();

  const int T = NoutPad >> 4;
  const int ct0 = wave << 1, ct1 = ct0 | 1;
  const bf16* Af = As + lane * 16;
  v8f acc0 = {}, acc1 = {};
  if (ct1 < T) {                       // scalar condition, hoisted out of loop
    const bf16* B0 = Bt + (size_t)ct0 * 16 * Kpad;
    const bf16* B1 = Bt + (size_t)ct1 * 16 * Kpad;
    v16bf b0 = load_b_frag(B0, Kpad, lane, 0);
    v16bf b1 = load_b_frag(B1, Kpad, lane, 0);
#pragma unroll 1
    for (int k0 = 0; k0 < Kpad; k0 += 32) {
      v16bf a = *(const v16bf*)(Af + ((k0 >> 5) << 9));
      v16bf nb0 = load_b_frag(B0, Kpad, lane, k0 + 32);  // last iter over-read
      v16bf nb1 = load_b_frag(B1, Kpad, lane, k0 + 32);  // lands in alloc pad
      acc0 = wmma_bf16(a, b0, acc0);
      acc1 = wmma_bf16(a, b1, acc1);
      b0 = nb0; b1 = nb1;
    }
  } else if (ct0 < T) {
    const bf16* B0 = Bt + (size_t)ct0 * 16 * Kpad;
#pragma unroll 1
    for (int k0 = 0; k0 < Kpad; k0 += 32) {
      v16bf a = *(const v16bf*)(Af + ((k0 >> 5) << 9));
      v16bf b = load_b_frag(B0, Kpad, lane, k0);
      acc0 = wmma_bf16(a, b, acc0);
    }
  }

  const int rb = (lane >> 4) << 3, cl = lane & 15;
  for (int j = 0; j < 2; ++j) {
    int ct = (wave << 1) | j;
    if (ct >= T) continue;
    int col = (ct << 4) + cl;
    if (col >= ldc) continue;
    float bv = bias ? bias[col] : 0.f;
    const v8f& acc = j ? acc1 : acc0;
#pragma unroll
    for (int r = 0; r < 8; ++r) {
      int row = m0 + rb + r;
      if (row < n) {
        float v = acc[r] + bv;
        if (relu) v = fmaxf(v, 0.f);
        C[(size_t)row * ldc + col] = v;
      }
    }
  }
}

// ---------------------------------------------------------------------------
// Fused CGL edge-message GEMM + scatter-add:
//   m[e] = relu(Wm * [x_bf16[src[e]] ; eattr[e]] + bm); agg[dst[e]] += m[e]
// block = 256 threads = 8 waves; tile = 16 edges x 256 cols; A pre-swizzled
// ---------------------------------------------------------------------------
__global__ void gnn_edge_msg(const bf16* __restrict__ xbf, int Kx, int xPitch,
                             const float* __restrict__ eattr,
                             const int* __restrict__ src, const int* __restrict__ dst,
                             int ne, const bf16* __restrict__ Wmt, int Kpad,
                             const float* __restrict__ bm, float* __restrict__ agg) {
  __shared__ __attribute__((aligned(32))) bf16 As[16 * 288];  // Kpad <= 288
  const int tid = threadIdx.x, lane = tid & 31;
  const int wave = __builtin_amdgcn_readfirstlane(tid >> 5);
  const int e0 = blockIdx.x * 16;
  const int tot = 16 * Kpad;
  for (int i = tid; i < tot; i += blockDim.x) {
    int r = i / Kpad, k = i - r * Kpad;
    int e = e0 + r;
    bf16 v = (bf16)0.f;
    if (e < ne) {
      if (k < Kx)          v = xbf[(size_t)src[e] * xPitch + k];
      else if (k < Kx + 3) v = (bf16)eattr[(size_t)e * 3 + (k - Kx)];
    }
    int kb = k >> 5, kr = k & 31;
    int l  = r + ((kr >> 4) << 4);
    int el = kr & 15;
    As[kb * 512 + l * 16 + el] = v;
  }
  __syncthreads();

  const int ct0 = wave << 1, ct1 = ct0 | 1;
  const bf16* B0 = Wmt + (size_t)ct0 * 16 * Kpad;
  const bf16* B1 = Wmt + (size_t)ct1 * 16 * Kpad;
  const bf16* Af = As + lane * 16;
  v8f acc0 = {}, acc1 = {};
  v16bf b0 = load_b_frag(B0, Kpad, lane, 0);
  v16bf b1 = load_b_frag(B1, Kpad, lane, 0);
#pragma unroll 1
  for (int k0 = 0; k0 < Kpad; k0 += 32) {
    v16bf a = *(const v16bf*)(Af + ((k0 >> 5) << 9));
    v16bf nb0 = load_b_frag(B0, Kpad, lane, k0 + 32);  // padded over-read
    v16bf nb1 = load_b_frag(B1, Kpad, lane, k0 + 32);
    acc0 = wmma_bf16(a, b0, acc0);
    acc1 = wmma_bf16(a, b1, acc1);
    b0 = nb0; b1 = nb1;
  }

  const int rb = (lane >> 4) << 3, cl = lane & 15;
  for (int j = 0; j < 2; ++j) {
    int col = ((((wave << 1) | j)) << 4) + cl;
    float bv = bm[col];
    const v8f& acc = j ? acc1 : acc0;
#pragma unroll
    for (int r = 0; r < 8; ++r) {
      int e = e0 + rb + r;
      if (e < ne) {
        float v = fmaxf(acc[r] + bv, 0.f);
        atomicAdd(&agg[(size_t)dst[e] * 256 + col], v);
      }
    }
  }
}

// ---------------------------------------------------------------------------
// GAT pieces
// ---------------------------------------------------------------------------
__global__ void gnn_gat_dots(const float* __restrict__ h, int n, int C,
                             const float* __restrict__ a_s, const float* __restrict__ a_d,
                             float* __restrict__ es, float* __restrict__ ed) {
  int i = blockIdx.x * blockDim.x + threadIdx.x;
  if (i >= n) return;
  float s = 0.f, d = 0.f;
  for (int c = 0; c < C; ++c) {
    float hv = h[(size_t)i * C + c];
    s += hv * a_s[c]; d += hv * a_d[c];
  }
  es[i] = s; ed[i] = d;
}

__global__ void gnn_gat_logit(const int* __restrict__ src, const int* __restrict__ dst,
                              const int* __restrict__ valid, int E_, int n,
                              const float* __restrict__ es, const float* __restrict__ ed,
                              float* __restrict__ logit, float* __restrict__ mmax) {
  int e = blockIdx.x * blockDim.x + threadIdx.x;
  int ne = E_ + n;
  if (e >= ne) return;
  int s, d, v;
  if (e < E_) { s = src[e]; d = dst[e]; v = valid[e]; }
  else        { s = d = e - E_; v = 1; }
  float z = es[s] + ed[d];
  z = z > 0.f ? z : 0.2f * z;   // leaky_relu 0.2
  logit[e] = z;
  if (v) atomicMaxF(&mmax[d], z);
}

__global__ void gnn_gat_exp(const int* __restrict__ src, const int* __restrict__ dst,
                            const int* __restrict__ valid, int E_, int n,
                            const float* __restrict__ logit, const float* __restrict__ mmax,
                            float* __restrict__ eexp, float* __restrict__ ssum) {
  int e = blockIdx.x * blockDim.x + threadIdx.x;
  int ne = E_ + n;
  if (e >= ne) return;
  int d, v;
  if (e < E_) { d = dst[e]; v = valid[e]; }
  else        { d = e - E_; v = 1; }
  float ex = 0.f;
  if (v) ex = expf(logit[e] - mmax[d]);
  eexp[e] = ex;
  if (v) atomicAdd(&ssum[d], ex);
}

__global__ void gnn_gat_agg(const int* __restrict__ src, const int* __restrict__ dst,
                            const int* __restrict__ valid, int E_, int n,
                            const float* __restrict__ eexp, const float* __restrict__ ssum,
                            const float* __restrict__ h, int C, float* __restrict__ out) {
  int nw = (gridDim.x * blockDim.x) >> 5;
  int w = (blockIdx.x * blockDim.x + threadIdx.x) >> 5;
  int lane = threadIdx.x & 31;
  int ne = E_ + n;
  for (int e = w; e < ne; e += nw) {
    int s, d, v;
    if (e < E_) { s = src[e]; d = dst[e]; v = valid[e]; }
    else        { s = d = e - E_; v = 1; }
    if (!v) continue;
    float coef = eexp[e] / (ssum[d] + 1e-16f);
    if (coef == 0.f) continue;
    for (int c = lane; c < C; c += 32)
      atomicAdd(&out[(size_t)d * C + c], coef * h[(size_t)s * C + c]);
  }
}

__global__ void gnn_bias_act(float* __restrict__ out, const float* __restrict__ b,
                             int n, int C, int relu) {
  int idx = blockIdx.x * blockDim.x + threadIdx.x;
  if (idx >= n * C) return;
  int c = idx % C;
  float v = out[idx] + b[c];
  if (relu) v = fmaxf(v, 0.f);
  out[idx] = v;
}

// ---------------------------------------------------------------------------
// top-k pool pieces
// ---------------------------------------------------------------------------
__global__ void gnn_vecnorm(const float* __restrict__ w, int C, float* __restrict__ out) {
  __shared__ float red[256];
  float s = 0.f;
  for (int c = threadIdx.x; c < C; c += blockDim.x) s += w[c] * w[c];
  red[threadIdx.x] = s;
  __syncthreads();
  for (int off = 128; off > 0; off >>= 1) {
    if ((int)threadIdx.x < off) red[threadIdx.x] += red[threadIdx.x + off];
    __syncthreads();
  }
  if (threadIdx.x == 0) out[0] = sqrtf(red[0]);
}

__global__ void gnn_score(const float* __restrict__ x, const float* __restrict__ w,
                          const float* __restrict__ nrm, int n, float* __restrict__ score) {
  int i = blockIdx.x * blockDim.x + threadIdx.x;
  if (i >= n) return;
  float s = 0.f;
  for (int c = 0; c < 256; ++c) s += x[(size_t)i * 256 + c] * w[c];
  score[i] = s / (nrm[0] + 1e-16f);
}

// stable descending rank (matches jax.lax.top_k tie-breaking)
__global__ void gnn_rank(const float* __restrict__ score, int n, int k,
                         int* __restrict__ perm, int* __restrict__ mapping) {
  int i = blockIdx.x * blockDim.x + threadIdx.x;
  if (i >= n) return;
  float si = score[i];
  int r = 0;
  for (int j = 0; j < n; ++j) {
    float sj = score[j];
    r += (sj > si) || (sj == si && j < i);
  }
  if (r < k) { perm[r] = i; mapping[i] = r; }
  else mapping[i] = -1;
}

__global__ void gnn_pool_gather(const float* __restrict__ x, const float* __restrict__ score,
                                const int* __restrict__ perm, int k, int C,
                                float* __restrict__ out) {
  int idx = blockIdx.x * blockDim.x + threadIdx.x;
  if (idx >= k * C) return;
  int j = idx / C, c = idx - j * C;
  int p = perm[j];
  out[idx] = x[(size_t)p * C + c] * tanhf(score[p]);
}

__global__ void gnn_remap(const int* __restrict__ src, const int* __restrict__ dst,
                          const int* __restrict__ valid, const int* __restrict__ mapping,
                          int E_, int* __restrict__ srcO, int* __restrict__ dstO,
                          int* __restrict__ validO) {
  int e = blockIdx.x * blockDim.x + threadIdx.x;
  if (e >= E_) return;
  int ns = mapping[src[e]], nd = mapping[dst[e]];
  int v = valid[e] && ns >= 0 && nd >= 0;
  srcO[e] = v ? ns : 0;
  dstO[e] = v ? nd : 0;
  validO[e] = v;
}

__global__ void gnn_copy(const float* __restrict__ src, float* __restrict__ dst, int n) {
  int i = blockIdx.x * blockDim.x + threadIdx.x;
  if (i < n) dst[i] = src[i];
}

__global__ void gnn_scatter_add(const float* __restrict__ small, const int* __restrict__ perm,
                                int k, int C, float* __restrict__ out) {
  int idx = blockIdx.x * blockDim.x + threadIdx.x;
  if (idx >= k * C) return;
  int j = idx / C, c = idx - j * C;
  out[(size_t)perm[j] * C + c] += small[idx];
}

__global__ void gnn_fill_f32(float* p, int n, float v) {
  int i = blockIdx.x * blockDim.x + threadIdx.x;
  if (i < n) p[i] = v;
}
__global__ void gnn_fill_i32(int* p, int n, int v) {
  int i = blockIdx.x * blockDim.x + threadIdx.x;
  if (i < n) p[i] = v;
}

// ---------------------------------------------------------------------------
// Host orchestration (graph-capture safe: kernels + hipMemsetAsync only)
// ---------------------------------------------------------------------------
extern "C" void kernel_launch(void* const* d_in, const int* in_sizes, int n_in,
                              void* d_out, int out_size, void* d_ws, size_t ws_size,
                              hipStream_t stream) {
  (void)in_sizes; (void)n_in; (void)out_size; (void)ws_size;
  const int N0 = 20000, E = 320000;
  const int n1 = 10000, n2 = 5000, n3 = 2500;

  // ---- flattened input map (setup_inputs dict order, recursive) ----
  const float* x_in  = (const float*)d_in[0];
  const int*   eidx  = (const int*)  d_in[1];
  const float* eattr = (const float*)d_in[2];
  const float* Wm0 = (const float*)d_in[3];  const float* bm0 = (const float*)d_in[4];
  const float* Wn0 = (const float*)d_in[5];  const float* bn0 = (const float*)d_in[6];
  const float* Wm1 = (const float*)d_in[7];  const float* bm1 = (const float*)d_in[8];
  const float* Wn1 = (const float*)d_in[9];  const float* bn1 = (const float*)d_in[10];
  const float* Wm2 = (const float*)d_in[11]; const float* bm2 = (const float*)d_in[12];
  const float* Wn2 = (const float*)d_in[13]; const float* bn2 = (const float*)d_in[14];
  const float* poolw[3] = {(const float*)d_in[15], (const float*)d_in[16],
                           (const float*)d_in[17]};
  struct GatP { const float *W, *as, *ad, *b; };
  GatP gd[3], gu[3];
  for (int i = 0; i < 3; ++i)
    gd[i] = {(const float*)d_in[18 + 4 * i], (const float*)d_in[19 + 4 * i],
             (const float*)d_in[20 + 4 * i], (const float*)d_in[21 + 4 * i]};
  for (int i = 0; i < 3; ++i)
    gu[i] = {(const float*)d_in[30 + 4 * i], (const float*)d_in[31 + 4 * i],
             (const float*)d_in[32 + 4 * i], (const float*)d_in[33 + 4 * i]};
  const int* src0 = eidx;
  const int* dst0 = eidx + E;

  // ---- bump allocator over d_ws ----
  char* wsp = (char*)d_ws;
  size_t used = 0;
  auto alloc = [&](size_t bytes) -> void* {
    void* p = wsp + used;
    used += (bytes + 255) & ~(size_t)255;
    return p;
  };
  auto cdiv = [](int a, int b) { return (a + b - 1) / b; };
  auto zero = [&](void* p, size_t bytes) { (void)hipMemsetAsync(p, 0, bytes, stream); };

  auto convW = [&](const float* W, int K, int N, int Kpad, int Npad) -> bf16* {
    // +128 elems slack so the B-fragment prefetch over-read stays in-bounds
    bf16* out = (bf16*)alloc(((size_t)Npad * Kpad + 128) * sizeof(bf16));
    int tot = Npad * Kpad;
    gnn_convW<<<cdiv(tot, 256), 256, 0, stream>>>(W, K, N, Kpad, out, tot);
    return out;
  };

  // weights -> transposed/padded bf16
  bf16* Wm0t = convW(Wm0, 11, 256, 32, 256);
  bf16* Wn0t = convW(Wn0, 264, 256, 288, 256);
  bf16* Wm1t = convW(Wm1, 259, 256, 288, 256);
  bf16* Wn1t = convW(Wn1, 512, 256, 512, 256);
  bf16* Wm2t = convW(Wm2, 259, 256, 288, 256);
  bf16* Wn2t = convW(Wn2, 512, 256, 512, 256);
  bf16* Wdt[3]; for (int i = 0; i < 3; ++i) Wdt[i] = convW(gd[i].W, 256, 256, 256, 256);
  bf16* Wut[3];
  Wut[0] = convW(gu[0].W, 256, 256, 256, 256);
  Wut[1] = convW(gu[1].W, 256, 256, 256, 256);
  Wut[2] = convW(gu[2].W, 256, 8, 256, 16);

  // buffers
  float* xa   = (float*)alloc((size_t)N0 * 256 * 4);     // xs0 after down stack
  float* xb   = (float*)alloc((size_t)N0 * 256 * 4);
  float* xc   = (float*)alloc((size_t)N0 * 256 * 4);
  float* hbuf = (float*)alloc((size_t)N0 * 256 * 4);     // GAT h / CGL agg (aliased)
  float* agg  = hbuf;
  float* xs1  = (float*)alloc((size_t)n1 * 256 * 4);
  float* xs2  = (float*)alloc((size_t)5008 * 256 * 4);
  bf16*  xbf  = (bf16*)alloc((size_t)N0 * 256 * 2);
  bf16*  abuf = (bf16*)alloc((size_t)N0 * 512 * 2);
  int *src1 = (int*)alloc(E * 4), *dst1 = (int*)alloc(E * 4), *val1 = (int*)alloc(E * 4);
  int *src2 = (int*)alloc(E * 4), *dst2 = (int*)alloc(E * 4), *val2 = (int*)alloc(E * 4);
  int *src3 = (int*)alloc(E * 4), *dst3 = (int*)alloc(E * 4), *val3 = (int*)alloc(E * 4);
  int *val0 = (int*)alloc(E * 4);
  int *p1 = (int*)alloc(n1 * 4), *p2 = (int*)alloc(n2 * 4), *p3 = (int*)alloc(n3 * 4);
  int *mapping = (int*)alloc(N0 * 4);
  float* score = (float*)alloc(N0 * 4);
  float* wnorm = (float*)alloc(256);
  float* es = (float*)alloc(N0 * 4);
  float* ed = (float*)alloc(N0 * 4);
  float* mmax = (float*)alloc(N0 * 4);
  float* ssum = (float*)alloc(N0 * 4);
  float* logit = (float*)alloc((size_t)(E + N0) * 4);
  float* eexp  = (float*)alloc((size_t)(E + N0) * 4);

  gnn_fill_i32<<<cdiv(E, 256), 256, 0, stream>>>(val0, E, 1);

  // ---- composite launch helpers ----
  auto pack = [&](const float* X1, int K1, const float* X2, int K2,
                  int n, int nPad, int Kpad, bf16* out) {
    int tot = nPad * Kpad;
    gnn_pack<<<cdiv(tot, 256), 256, 0, stream>>>(X1, K1, X2, K2, n, Kpad, out, tot);
  };
  auto gemm = [&](const bf16* A, int nPad, int Kpad, const bf16* Bt, int NoutPad,
                  const float* bias, float* C, int n, int ldc, int relu) {
    gnn_gemm<<<nPad / 16, 256, 0, stream>>>(A, Kpad, Bt, NoutPad, bias, C, n, ldc, relu);
  };

  auto cgl = [&](const float* x, int Kx, const bf16* Wmt, int KmPad, const float* bm,
                 const bf16* Wnt, int KnPad, const float* bn, float* out, int relu) {
    int xPitch = (Kx == 8) ? 8 : 256;
    pack(x, Kx, nullptr, 0, N0, N0, xPitch, xbf);
    zero(agg, (size_t)N0 * 256 * 4);
    gnn_edge_msg<<<cdiv(E, 16), 256, 0, stream>>>(xbf, Kx, xPitch, eattr, src0, dst0,
                                                  E, Wmt, KmPad, bm, agg);
    pack(x, Kx, agg, 256, N0, N0, KnPad, abuf);
    gemm(abuf, N0, KnPad, Wnt, 256, bn, out, N0, 256, relu);
  };

  auto gat = [&](const float* x, int n, const int* src, const int* dst, const int* valid,
                 const bf16* Wt, int NoutPad, int outC,
                 const float* a_s, const float* a_d, const float* b,
                 float* out, int relu) {
    int nPad = (n + 15) & ~15;
    pack(x, 256, nullptr, 0, n, nPad, 256, abuf);
    gemm(abuf, nPad, 256, Wt, NoutPad, nullptr, hbuf, n, outC, 0);
    gnn_gat_dots<<<cdiv(n, 256), 256, 0, stream>>>(hbuf, n, outC, a_s, a_d, es, ed);
    gnn_fill_f32<<<cdiv(n, 256), 256, 0, stream>>>(mmax, n, -1e30f);
    zero(ssum, (size_t)n * 4);
    int ne = E + n;
    gnn_gat_logit<<<cdiv(ne, 256), 256, 0, stream>>>(src, dst, valid, E, n, es, ed,
                                                     logit, mmax);
    gnn_gat_exp<<<cdiv(ne, 256), 256, 0, stream>>>(src, dst, valid, E, n, logit, mmax,
                                                   eexp, ssum);
    zero(out, (size_t)n * outC * 4);
    gnn_gat_agg<<<2048, 256, 0, stream>>>(src, dst, valid, E, n, eexp, ssum, hbuf,
                                          outC, out);
    gnn_bias_act<<<cdiv(n * outC, 256), 256, 0, stream>>>(out, b, n, outC, relu);
  };

  auto pool = [&](const float* x, int n, const float* w, int k,
                  const int* srcI, const int* dstI, const int* valI,
                  float* xout, int* srcO, int* dstO, int* valO, int* perm) {
    gnn_vecnorm<<<1, 256, 0, stream>>>(w, 256, wnorm);
    gnn_score<<<cdiv(n, 256), 256, 0, stream>>>(x, w, wnorm, n, score);
    gnn_rank<<<cdiv(n, 256), 256, 0, stream>>>(score, n, k, perm, mapping);
    gnn_pool_gather<<<cdiv(k * 256, 256), 256, 0, stream>>>(x, score, perm, k, 256, xout);
    gnn_remap<<<cdiv(E, 256), 256, 0, stream>>>(srcI, dstI, valI, mapping, E,
                                                srcO, dstO, valO);
  };

  auto unpool = [&](const float* res, const float* small, const int* perm,
                    int k, int n, float* out) {
    gnn_copy<<<cdiv(n * 256, 256), 256, 0, stream>>>(res, out, n * 256);
    gnn_scatter_add<<<cdiv(k * 256, 256), 256, 0, stream>>>(small, perm, k, 256, out);
  };

  // ---- forward pipeline ----
  // CGL stack (relu fused into down2 node update)
  cgl(x_in, 8,   Wm0t, 32,  bm0, Wn0t, 288, bn0, xa, 0);
  cgl(xa,   256, Wm1t, 288, bm1, Wn1t, 512, bn1, xb, 0);
  cgl(xb,   256, Wm2t, 288, bm2, Wn2t, 512, bn2, xa, 1);   // xa == xs[0]

  // Down path
  pool(xa, N0, poolw[0], n1, src0, dst0, val0, xb, src1, dst1, val1, p1);
  gat(xb, n1, src1, dst1, val1, Wdt[0], 256, 256, gd[0].as, gd[0].ad, gd[0].b, xs1, 1);
  pool(xs1, n1, poolw[1], n2, src1, dst1, val1, xb, src2, dst2, val2, p2);
  gat(xb, n2, src2, dst2, val2, Wdt[1], 256, 256, gd[1].as, gd[1].ad, gd[1].b, xs2, 1);
  pool(xs2, n2, poolw[2], n3, src2, dst2, val2, xb, src3, dst3, val3, p3);
  gat(xb, n3, src3, dst3, val3, Wdt[2], 256, 256, gd[2].as, gd[2].ad, gd[2].b, xc, 1);

  // Up path
  unpool(xs2, xc, p3, n3, n2, xb);
  gat(xb, n2, src2, dst2, val2, Wut[0], 256, 256, gu[0].as, gu[0].ad, gu[0].b, xc, 1);
  unpool(xs1, xc, p2, n2, n1, xb);
  gat(xb, n1, src1, dst1, val1, Wut[1], 256, 256, gu[1].as, gu[1].ad, gu[1].b, xc, 1);
  unpool(xa, xc, p1, n1, N0, xb);
  gat(xb, N0, src0, dst0, val0, Wut[2], 16, 8, gu[2].as, gu[2].ad, gu[2].b,
      (float*)d_out, 0);
}
